// R22SDF_89060441849982
// MI455X (gfx1250) — matrix operand order repaired and checked by reading, not compiled, not run on hardware
//
#include <hip/hip_runtime.h>

typedef __attribute__((ext_vector_type(16))) _Float16 v16h;
typedef __attribute__((ext_vector_type(8)))  float    v8f;

#define NREAL 512
#define WAVES_PER_BLOCK 8
#define ROWS_PER_WAVE 4
#define SLICE 576   // floats of LDS per wave (staging 512 / transpose 544 / untangle 512)

// ---- fixed-point helpers (match reference _fxp/_quant semantics) ----
__device__ __forceinline__ float fxp_floor(float x, float s, float lo, float hi) {
    float q = floorf(x * s) / s;
    return fminf(fmaxf(q, lo), hi);
}
__device__ __forceinline__ float fxp_round(float x, float s, float lo, float hi) {
    float q = rintf(x * s) / s;   // round-half-even, same as np.round
    return fminf(fmaxf(q, lo), hi);
}

// v_cos_f32 / v_sin_f32 take input in REVOLUTIONS (hw computes cos(2*pi*t)).
// All angles here are exact multiples of 1/512 of a circle and results are
// quantized to >=1/32 steps, so single-instruction hw sin/cos suffices.
__device__ __forceinline__ float hw_cos_rev(float t) { return __builtin_amdgcn_cosf(t); }
__device__ __forceinline__ float hw_sin_rev(float t) { return __builtin_amdgcn_sinf(t); }

// quantized twiddle, WN_COEFF=(6,4) nearest, reference _w_N special cases,
// fully BRANCHLESS (selects, no exec-mask divergence).
__device__ __forceinline__ void wq64(float t, float& wr, float& wi) {
    float cf = fxp_round(hw_cos_rev(t), 16.f, -2.f, 1.9375f);
    float sf = fxp_round(hw_sin_rev(t), 16.f, -2.f, 1.9375f);
    bool cpm = (fabsf(cf) == 1.f);      // cf == +/-1  -> (cf, 0)
    bool spm = (fabsf(sf) == 1.f);      // sf == +/-1  -> (0, sf)   (lower priority)
    wr = (!cpm && spm) ? 0.f : cf;
    wi = cpm ? 0.f : sf;
}

// intra-wave LDS phase ordering: each wave owns its LDS slice and CDNA5 LDS
// ops from one wave complete in order, so a compiler-level fence is enough.
__device__ __forceinline__ void wave_lds_fence() {
    __builtin_amdgcn_fence(__ATOMIC_ACQ_REL, "wavefront");
}

__global__ __launch_bounds__(256)
void R22SDF_wmma_fft512(const float* __restrict__ xin,
                        float* __restrict__ out, int rows)
{
    __shared__ float smem[WAVES_PER_BLOCK * SLICE];
    const int  lane   = threadIdx.x & 31;
    const int  wave   = threadIdx.x >> 5;
    const bool hiLane = lane >= 16;
    const int  nIdx   = lane & 15;
    float* xs = &smem[wave * SLICE];
    const int  waveG  = blockIdx.x * WAVES_PER_BLOCK + wave;
    const int  row0   = waveG * ROWS_PER_WAVE;

    // =============== row-invariant precompute (hoisted out of row loop) ======
    // DFT16 A-operands, complex packed over K: A1=[Wr|-Wi] (real out),
    // A2=[Wi|Wr] (imag out).  K order used here (contiguous, B matches):
//   lane<16 : elem j -> K = j
//   lane>=16: elem j -> K = j+16
    v16h A1, A2;
    {
        const int m = nIdx;
        #pragma unroll
        for (int j = 0; j < 16; ++j) {
            int K  = hiLane ? j + 16
                            : j;
            int n2 = K & 15;
            bool imagHalf = (K >= 16);
            float trev = -(float)((m * n2) & 15) * (1.0f / 16.0f);
            float wr, wi; wq64(trev, wr, wi);
            A1[j] = (_Float16)(imagHalf ? -wi : wr);
            A2[j] = (_Float16)(imagHalf ?  wr : wi);
        }
    }
    // inter-pass twiddles W256^(n1*k2), per C/D row r (M = r + 8*hiLane)
    float tw_r[8], tw_i[8];
    #pragma unroll
    for (int r = 0; r < 8; ++r) {
        int k2 = r + (hiLane ? 8 : 0);
        wq64(-(float)((k2 * nIdx) & 255) * (1.0f / 256.0f), tw_r[r], tw_i[r]);
    }
    // EXTRA-stage coeffs (4,2): exp(-i*pi*k/256), k = lane + 32*it
    float ex_r[9], ex_i[9];
    #pragma unroll
    for (int it = 0; it < 9; ++it) {
        float trev = (float)(lane + 32 * it) * (1.0f / 512.0f);
        ex_r[it] = fxp_round( hw_cos_rev(trev), 4.f, -2.f, 1.75f);
        ex_i[it] = fxp_round(-hw_sin_rev(trev), 4.f, -2.f, 1.75f);
    }
    const v8f c0 = {};

    // ============================== row loop =================================
    for (int rr = 0; rr < ROWS_PER_WAVE; ++rr) {
        const int row = row0 + rr;
        if (row >= rows) break;

        // ---- stage the input row into this wave's LDS slice ----
        {
            const float* src = xin + (size_t)row * NREAL;
            #pragma unroll
            for (int t = 0; t < 4; ++t) {
                float4 v = *(const float4*)(src + lane * 16 + t * 4);
                *(float4*)(xs + lane * 16 + t * 4) = v;
            }
            if (row + 1 < rows)   // CDNA5 global_prefetch_b8 of next row
                __builtin_prefetch(xin + (size_t)(row + 1) * NREAL + lane * 16, 0, 0);
        }
        wave_lds_fence();

        // ---- B1: Zr rows K=0..15 (lanes 0-15), Zi rows K=16..31 ----
        v16h B1;
        #pragma unroll
        for (int j = 0; j < 16; ++j) {
            float v = hiLane ? xs[2 * (nIdx + 16 * j) + 1]
                             : xs[2 * (nIdx + 16 * j)];
            B1[j] = (_Float16)(v * 16.0f);   // PRE_GAIN
        }

        // ---- pass 1: DFT16 over n2 (one WMMA per complex component) ----
        v8f Dr = __builtin_amdgcn_wmma_f32_16x16x32_f16(false, A1, false, B1, (short)0, c0, false, false);
        v8f Di = __builtin_amdgcn_wmma_f32_16x16x32_f16(false, A2, false, B1, (short)0, c0, false, false);
        wave_lds_fence();   // xs reused as transpose buffer

        // ---- twiddle, quantize (8,5), transpose via stride-17 LDS rows ----
        #pragma unroll
        for (int r = 0; r < 8; ++r) {
            int k2 = r + (hiLane ? 8 : 0);
            float tr = Dr[r] * tw_r[r] - Di[r] * tw_i[r];
            float ti = Dr[r] * tw_i[r] + Di[r] * tw_r[r];
            tr = fxp_floor(tr, 32.f, -4.f, 4.f - 0.03125f);
            ti = fxp_floor(ti, 32.f, -4.f, 4.f - 0.03125f);
            xs[      k2 * 17 + nIdx] = tr;
            xs[272 + k2 * 17 + nIdx] = ti;
        }
        wave_lds_fence();

        // ---- B2 = T^T : Tr rows K=0..15, Ti rows K=16..31 ----
        v16h B2;
        #pragma unroll
        for (int j = 0; j < 16; ++j) {
            float v = hiLane ? xs[272 + nIdx * 17 + j]
                             : xs[      nIdx * 17 + j];
            B2[j] = (_Float16)v;
        }

        // ---- pass 2: DFT16 over n1 ----
        v8f Xr = __builtin_amdgcn_wmma_f32_16x16x32_f16(false, A1, false, B2, (short)0, c0, false, false);
        v8f Xi = __builtin_amdgcn_wmma_f32_16x16x32_f16(false, A2, false, B2, (short)0, c0, false, false);
        wave_lds_fence();   // xs reused as spectrum buffer

        // ---- store X[k], k = k2 + 16*k1 (already natural order) ----
        #pragma unroll
        for (int r = 0; r < 8; ++r) {
            int k = nIdx + 16 * (r + (hiLane ? 8 : 0));
            xs[2 * k]     = Xr[r];
            xs[2 * k + 1] = Xi[r];
        }
        wave_lds_fence();

        // ---- real-FFT untangle + EXTRA + final quant (8,2) ----
        {
            float* orow = out + (size_t)row * 257 * 2;
            #pragma unroll
            for (int it = 0; it < 9; ++it) {
                int k = lane + 32 * it;
                if (k <= 256) {
                    int km = k & 255;
                    int kr = (256 - km) & 255;
                    float zre = xs[2 * km], zim =  xs[2 * km + 1];
                    float rre = xs[2 * kr], rim = -xs[2 * kr + 1];   // conj
                    float xer  = 0.5f * (zre + rre), xei =  0.5f * (zim + rim);
                    float xor_ = 0.5f * (zim - rim), xoi = -0.5f * (zre - rre);
                    float ore = xer + ex_r[it] * xor_ - ex_i[it] * xoi;
                    float oim = xei + ex_r[it] * xoi + ex_i[it] * xor_;
                    float2 o;
                    o.x = fxp_floor(ore, 4.f, -32.f, 31.75f) * (1.0f / 16.0f);
                    o.y = fxp_floor(oim, 4.f, -32.f, 31.75f) * (1.0f / 16.0f);
                    *(float2*)(orow + 2 * k) = o;
                }
            }
        }
        wave_lds_fence();   // WAR: next iteration re-stages into xs
    }
}

extern "C" void kernel_launch(void* const* d_in, const int* in_sizes, int n_in,
                              void* d_out, int out_size, void* d_ws, size_t ws_size,
                              hipStream_t stream) {
    (void)n_in; (void)out_size; (void)d_ws; (void)ws_size;
    const float* x = (const float*)d_in[0];
    float* out = (float*)d_out;
    int rows = in_sizes[0] / NREAL;                         // 16384
    int rowsPerBlock = WAVES_PER_BLOCK * ROWS_PER_WAVE;     // 32
    int blocks = (rows + rowsPerBlock - 1) / rowsPerBlock;  // 512
    hipLaunchKernelGGL(R22SDF_wmma_fft512, dim3(blocks), dim3(256), 0, stream,
                       x, out, rows);
}
